// PoseCNN_55740085567522
// MI455X (gfx1250) — compile-verified, matrix operand order checked by reading, not run on hardware
//
#include <hip/hip_runtime.h>
#include <hip/hip_bf16.h>

// ---------------------------------------------------------------------------
// PoseCNN pipeline for MI455X (gfx1250, wave32, WMMA, async LDS copies).
// Activations: NHWC _Float16 (halves HBM traffic, unit-stride channel dots,
// 16B-contiguous WMMA B-fragment half-rows -> global_load_async_to_lds_b128).
// ---------------------------------------------------------------------------

typedef __attribute__((ext_vector_type(16))) _Float16 v16h;
typedef __attribute__((ext_vector_type(8)))  _Float16 v8h;
typedef __attribute__((ext_vector_type(8)))  float    v8f;
typedef __attribute__((ext_vector_type(4)))  int      vi4;

#define NEGV (-1e30f)

#if __has_builtin(__builtin_amdgcn_global_load_async_to_lds_b128)
#define ASYNC_LDS 1
#else
#define ASYNC_LDS 0
#endif

__device__ __forceinline__ void wait_async_lds()
{
#if ASYNC_LDS
#if __has_builtin(__builtin_amdgcn_s_wait_asynccnt)
    __builtin_amdgcn_s_wait_asynccnt(0);
#else
    asm volatile("s_wait_asynccnt 0" ::: "memory");
#endif
#endif
}

__device__ __forceinline__ int imin(int a, int b) { return a < b ? a : b; }
__device__ __forceinline__ int imax(int a, int b) { return a > b ? a : b; }

// K-element mapping for the 16-bit 16x32 A/B fragments (ISA 7.12.2):
// lanes 0-15: v0..v3 hold K=0..7, v4..v7 hold K=16..23; lanes 16-31: +8.
__device__ __forceinline__ int kmap(int i, bool hi) {
    int v = i >> 1, w = i & 1;
    int k = (v < 4) ? ((v << 1) + w) : (16 + ((v - 4) << 1) + w);
    return k + (hi ? 8 : 0);
}

// ---------------------------------------------------------------------------
// Implicit-GEMM 3x3 conv + bias + ReLU, NHWC f16 in/out.
// Decomposed into 9 tap-GEMMs (no div/mod in the hot loop).
// Wave tile: 16 Cout x 32 pixels (two 16x16 WMMA tiles -> A reuse).
// Block: 256 threads = 8 waves => 128 Cout x 32 pixels; B tiles staged in LDS
// via GLOBAL_LOAD_ASYNC_TO_LDS_B128 (each half-row = 8 contiguous channels).
// grid: x = ceil(W/32), y = H, z = B * ceil(Cout/128)
// ---------------------------------------------------------------------------
__global__ __launch_bounds__(256) void conv3x3_wmma_kernel(
    const _Float16* __restrict__ in, const float* __restrict__ wgt,
    const float* __restrict__ bias, _Float16* __restrict__ out,
    int Cin, int Cpad, int Cout, int H, int W)
{
    const int coutTiles = (Cout + 127) >> 7;
    const int b    = blockIdx.z / coutTiles;
    const int ct   = blockIdx.z - b * coutTiles;
    const int y    = blockIdx.y;
    const int x0   = blockIdx.x << 5;
    const int tid  = threadIdx.x;
    const int lane = tid & 31;
    const int wave = tid >> 5;
    const int coutBase = (ct << 7) + (wave << 4);
    const bool active = (coutBase < Cout);
    const int m    = lane & 15;
    const bool hi  = lane >= 16;

    // Staging role: threads 0..127 each own one 8-channel half-row.
    const bool stager = tid < 128;
    const int s_tile = (tid >> 6) & 1;
    const int s_dl   = tid & 31;
    const int s_h    = (tid >> 5) & 1;
    const int s_n    = (s_tile << 4) + (s_dl & 15);
    const int s_chan = s_h * 16 + ((s_dl >= 16) ? 8 : 0);

    __shared__ __align__(16) _Float16 ldsB[2][32][16];

    v8f acc[2];
#pragma unroll
    for (int t2 = 0; t2 < 2; ++t2)
#pragma unroll
        for (int r = 0; r < 8; ++r) acc[t2][r] = 0.f;

    const float* wrow = wgt + (size_t)(coutBase + m) * Cin * 9;

    for (int tap = 0; tap < 9; ++tap) {
        const int dy = tap / 3 - 1;         // uniform (SALU)
        const int dx = tap - (tap / 3) * 3 - 1;
        const int yy = y + dy;
        const int xx = x0 + s_n + dx;
        const bool valid = stager && yy >= 0 && yy < H && xx >= 0 && xx < W;
        const _Float16* gpix =
            in + ((size_t)(b * H + yy) * W + xx) * Cpad + s_chan;
        _Float16* lrow = &ldsB[s_tile][s_dl][s_h * 8];

        for (int cb = 0; cb < Cpad; cb += 32) {
            // ---- stage B tiles (activations) ----
            if (valid) {
#if ASYNC_LDS
                __builtin_amdgcn_global_load_async_to_lds_b128(
                    (__attribute__((address_space(1))) vi4*)(void*)(gpix + cb),
                    (__attribute__((address_space(3))) vi4*)(void*)lrow, 0, 0);
#else
                *(v8h*)lrow = *(const v8h*)(gpix + cb);
#endif
            } else if (stager && cb == 0) {  // zero-fill once; rows stay zero
#pragma unroll
                for (int q = 0; q < 8; ++q) lrow[q] = (_Float16)0.f;
            }
            wait_async_lds();
            __syncthreads();

            v16h bv0 = *(const v16h*)(&ldsB[0][lane][0]);
            v16h bv1 = *(const v16h*)(&ldsB[1][lane][0]);

            if (active) {  // wave-uniform: EXEC all-1s inside (WMMA requirement)
                v16h av;
#pragma unroll
                for (int i = 0; i < 16; ++i) {
                    const int c = cb + kmap(i, hi);
                    av[i] = (c < Cin) ? (_Float16)wrow[c * 9 + tap]
                                      : (_Float16)0.f;
                }
                acc[0] = __builtin_amdgcn_wmma_f32_16x16x32_f16(
                    false, av, false, bv0, (short)0, acc[0], false, false);
                acc[1] = __builtin_amdgcn_wmma_f32_16x16x32_f16(
                    false, av, false, bv1, (short)0, acc[1], false, false);
            }
            __syncthreads();
        }
    }

    if (active) {
        const int cobase = coutBase + (hi ? 8 : 0);  // 8 consecutive Couts/lane
#pragma unroll
        for (int t2 = 0; t2 < 2; ++t2) {
            const int x = x0 + (t2 << 4) + (lane & 15);
            if (x < W) {
                v8h pk;
#pragma unroll
                for (int r = 0; r < 8; ++r)
                    pk[r] = (_Float16)fmaxf(acc[t2][r] + bias[cobase + r], 0.f);
                *(v8h*)(out + ((size_t)(b * H + y) * W + x) * Cout + cobase) = pk;
            }
        }
    }
}

// ---------------------------------------------------------------------------
// NCHW f32 -> NHWC f16 with channel padding (VGG input, 3 -> 32 channels)
// ---------------------------------------------------------------------------
__global__ void nchw2nhwc_pad_kernel(const float* __restrict__ in,
                                     _Float16* __restrict__ out,
                                     int C, int Cpad, int H, int W, int total)
{
    int idx = blockIdx.x * blockDim.x + threadIdx.x;
    if (idx >= total) return;                 // total = B*H*W
    int x = idx % W; int t = idx / W;
    int y = t % H;   int b = t / H;
    _Float16* op = out + (size_t)idx * Cpad;
    for (int c = 0; c < Cpad; ++c) {
        float v = (c < C) ? in[(((size_t)b * C + c) * H + y) * W + x] : 0.f;
        op[c] = (_Float16)v;
    }
}

// ---------------------------------------------------------------------------
// FC GEMM: out[M,N] = act(X[M,K] @ W[N,K]^T + b[N]).  One wave per 16x16 tile.
// ---------------------------------------------------------------------------
__global__ __launch_bounds__(32) void fc_wmma_kernel(
    const float* __restrict__ X, const float* __restrict__ Wt,
    const float* __restrict__ bias, float* __restrict__ out,
    int M, int N, int K, int doRelu)
{
    const int lane  = threadIdx.x & 31;
    const int nBase = blockIdx.x << 4;
    const int m     = lane & 15;
    const bool hi   = lane >= 16;
    const int nb    = nBase + m;

    v8f acc = {0.f, 0.f, 0.f, 0.f, 0.f, 0.f, 0.f, 0.f};

    for (int kb = 0; kb < K; kb += 32) {
        v16h av, bv;
#pragma unroll
        for (int i = 0; i < 16; ++i) {
            const int kk = kb + kmap(i, hi);
            av[i] = (m < M && kk < K) ? (_Float16)X[(size_t)m * K + kk] : (_Float16)0.f;
            bv[i] = (nb < N && kk < K) ? (_Float16)Wt[(size_t)nb * K + kk] : (_Float16)0.f;
        }
        acc = __builtin_amdgcn_wmma_f32_16x16x32_f16(
            false, av, false, bv, (short)0, acc, false, false);
    }

    const int nc = nBase + (lane & 15);
    if (nc < N) {
#pragma unroll
        for (int r = 0; r < 8; ++r) {
            const int mm = r + (hi ? 8 : 0);
            if (mm < M) {
                float v = acc[r] + bias[nc];
                out[(size_t)mm * N + nc] = doRelu ? fmaxf(v, 0.f) : v;
            }
        }
    }
}

// ---------------------------------------------------------------------------
// 2x2 max pool (stride 2), NHWC f16
// ---------------------------------------------------------------------------
__global__ void maxpool2_nhwc_kernel(const _Float16* __restrict__ in,
                                     _Float16* __restrict__ out,
                                     int C, int H, int W, int total)
{
    int idx = blockIdx.x * blockDim.x + threadIdx.x;
    if (idx >= total) return;                 // total = B*Ho*Wo*C
    const int Ho = H >> 1, Wo = W >> 1;
    int c = idx % C; int t = idx / C;
    int x = t % Wo;  t /= Wo;
    int y = t % Ho;  int b = t / Ho;
    const _Float16* p = in + ((size_t)(b * H + 2 * y) * W + 2 * x) * C + c;
    float v0 = (float)p[0],             v1 = (float)p[C];
    float v2 = (float)p[(size_t)W * C], v3 = (float)p[(size_t)W * C + C];
    out[idx] = (_Float16)fmaxf(fmaxf(v0, v1), fmaxf(v2, v3));
}

// ---------------------------------------------------------------------------
// 1x1 conv + ReLU: NHWC f16 in -> NCHW f32 out (unit-stride channel dot)
// ---------------------------------------------------------------------------
__global__ void conv1x1_relu_kernel(const _Float16* __restrict__ in,
                                    const float* __restrict__ w,
                                    const float* __restrict__ bias,
                                    float* __restrict__ out,
                                    int Cin, int Cout, int HW, int total)
{
    int idx = blockIdx.x * blockDim.x + threadIdx.x;
    if (idx >= total) return;
    int p = idx % HW; int t = idx / HW;
    int o = t % Cout; int b = t / Cout;
    const _Float16* ip = in + ((size_t)b * HW + p) * Cin;
    const float* wr = w + (size_t)o * Cin;
    float s = bias[o];
    for (int c = 0; c < Cin; ++c) s += (float)ip[c] * wr[c];
    out[idx] = fmaxf(s, 0.f);
}

// dst(B,C,2Hs,2Ws) += bilinear_upsample2x(src(B,C,Hs,Ws)), both NCHW f32
__global__ void up2add_kernel(const float* __restrict__ src, float* __restrict__ dst,
                              int C, int Hs, int Ws, int total)
{
    int idx = blockIdx.x * blockDim.x + threadIdx.x;
    if (idx >= total) return;
    const int Hd = Hs * 2, Wd = Ws * 2;
    int x = idx % Wd; int t = idx / Wd;
    int y = t % Hd;   t /= Hd;
    int c = t % C;    int b = t / C;
    float sy = (y + 0.5f) * 0.5f - 0.5f;
    float sx = (x + 0.5f) * 0.5f - 0.5f;
    float fy = floorf(sy), fx = floorf(sx);
    float wy = sy - fy, wx = sx - fx;
    int y0 = imin(imax((int)fy, 0), Hs - 1), y1 = imin(imax((int)fy + 1, 0), Hs - 1);
    int x0 = imin(imax((int)fx, 0), Ws - 1), x1 = imin(imax((int)fx + 1, 0), Ws - 1);
    const float* p = src + ((size_t)b * C + c) * Hs * Ws;
    float v = p[(size_t)y0 * Ws + x0] * (1.f - wy) * (1.f - wx)
            + p[(size_t)y0 * Ws + x1] * (1.f - wy) * wx
            + p[(size_t)y1 * Ws + x0] * wy * (1.f - wx)
            + p[(size_t)y1 * Ws + x1] * wy * wx;
    dst[idx] += v;
}

// ---------------------------------------------------------------------------
// Fused: upsample(seg_feat, 8x) -> 1x1 conv (11) -> relu -> softmax -> argmax
// ---------------------------------------------------------------------------
__global__ void seg_head_kernel(const float* __restrict__ feat,   // B x 64 x 60 x 80
                                const float* __restrict__ w,      // 11 x 64
                                const float* __restrict__ bias,   // 11
                                float* __restrict__ prob,         // B x 11 x 480 x 640
                                float* __restrict__ seg,          // B x 480 x 640
                                int total)
{
    int idx = blockIdx.x * blockDim.x + threadIdx.x;
    if (idx >= total) return;
    const int H = 480, W = 640, Hf = 60, Wf = 80, C = 64, NC = 11;
    int x = idx % W; int t = idx / W;
    int y = t % H;   int b = t / H;
    float sy = (y + 0.5f) * 0.125f - 0.5f;
    float sx = (x + 0.5f) * 0.125f - 0.5f;
    float fy = floorf(sy), fx = floorf(sx);
    float wy = sy - fy, wx = sx - fx;
    int y0 = imin(imax((int)fy, 0), Hf - 1), y1 = imin(imax((int)fy + 1, 0), Hf - 1);
    int x0 = imin(imax((int)fx, 0), Wf - 1), x1 = imin(imax((int)fx + 1, 0), Wf - 1);
    float w00 = (1.f - wy) * (1.f - wx), w01 = (1.f - wy) * wx;
    float w10 = wy * (1.f - wx),         w11 = wy * wx;

    float logit[NC];
#pragma unroll
    for (int k = 0; k < NC; ++k) logit[k] = bias[k];
    const float* base = feat + (size_t)b * C * Hf * Wf;
    for (int c = 0; c < C; ++c) {
        const float* p = base + (size_t)c * Hf * Wf;
        float s = w00 * p[(size_t)y0 * Wf + x0] + w01 * p[(size_t)y0 * Wf + x1]
                + w10 * p[(size_t)y1 * Wf + x0] + w11 * p[(size_t)y1 * Wf + x1];
#pragma unroll
        for (int k = 0; k < NC; ++k) logit[k] += s * w[k * C + c];
    }
    float mx = fmaxf(logit[0], 0.f);
    int arg = 0;
#pragma unroll
    for (int k = 0; k < NC; ++k) {
        logit[k] = fmaxf(logit[k], 0.f);
        if (logit[k] > mx) { mx = logit[k]; arg = k; }
    }
    float sum = 0.f;
    float e[NC];
#pragma unroll
    for (int k = 0; k < NC; ++k) { e[k] = __expf(logit[k] - mx); sum += e[k]; }
    float inv = 1.f / sum;
    size_t HW = (size_t)H * W;
#pragma unroll
    for (int k = 0; k < NC; ++k)
        prob[((size_t)b * NC + k) * HW + (size_t)y * W + x] = e[k] * inv;
    seg[(size_t)b * HW + (size_t)y * W + x] = (float)arg;
}

// ---------------------------------------------------------------------------
// Fused: upsample(tr_feat, 8x) -> 1x1 conv (30), no relu
// ---------------------------------------------------------------------------
__global__ void tr_head_kernel(const float* __restrict__ feat,   // B x 128 x 60 x 80
                               const float* __restrict__ w,      // 30 x 128
                               const float* __restrict__ bias,   // 30
                               float* __restrict__ trans,        // B x 30 x 480 x 640
                               int total)
{
    int idx = blockIdx.x * blockDim.x + threadIdx.x;
    if (idx >= total) return;
    const int H = 480, W = 640, Hf = 60, Wf = 80, C = 128, NO = 30;
    int x = idx % W; int t = idx / W;
    int y = t % H;   int b = t / H;
    float sy = (y + 0.5f) * 0.125f - 0.5f;
    float sx = (x + 0.5f) * 0.125f - 0.5f;
    float fy = floorf(sy), fx = floorf(sx);
    float wy = sy - fy, wx = sx - fx;
    int y0 = imin(imax((int)fy, 0), Hf - 1), y1 = imin(imax((int)fy + 1, 0), Hf - 1);
    int x0 = imin(imax((int)fx, 0), Wf - 1), x1 = imin(imax((int)fx + 1, 0), Wf - 1);
    float w00 = (1.f - wy) * (1.f - wx), w01 = (1.f - wy) * wx;
    float w10 = wy * (1.f - wx),         w11 = wy * wx;

    float accv[NO];
#pragma unroll
    for (int k = 0; k < NO; ++k) accv[k] = bias[k];
    const float* base = feat + (size_t)b * C * Hf * Wf;
    for (int c = 0; c < C; ++c) {
        const float* p = base + (size_t)c * Hf * Wf;
        float s = w00 * p[(size_t)y0 * Wf + x0] + w01 * p[(size_t)y0 * Wf + x1]
                + w10 * p[(size_t)y1 * Wf + x0] + w11 * p[(size_t)y1 * Wf + x1];
#pragma unroll
        for (int k = 0; k < NO; ++k) accv[k] += s * w[k * C + c];
    }
    size_t HW = (size_t)H * W;
#pragma unroll
    for (int k = 0; k < NO; ++k)
        trans[((size_t)b * NO + k) * HW + (size_t)y * W + x] = accv[k];
}

// ---------------------------------------------------------------------------
// label2bbx: init / atomic scan / finalize
// ---------------------------------------------------------------------------
__global__ void bbox_init_kernel(int* xs, int* ys, int* xe, int* ye, int* cnt)
{
    int i = threadIdx.x;
    if (i < 18) { xs[i] = 640; ys[i] = 480; xe[i] = -1; ye[i] = -1; cnt[i] = 0; }
}

__global__ void bbox_scan_kernel(const float* __restrict__ seg,
                                 int* xs, int* ys, int* xe, int* ye, int* cnt, int total)
{
    int idx = blockIdx.x * blockDim.x + threadIdx.x;
    if (idx >= total) return;
    const int W = 640, H = 480;
    int x = idx % W; int t = idx / W;
    int y = t % H;   int b = t / H;
    int c = (int)(seg[idx] + 0.5f);
    if (c >= 1 && c <= 9) {
        int q = b * 9 + (c - 1);
        atomicMin(&xs[q], x); atomicMax(&xe[q], x);
        atomicMin(&ys[q], y); atomicMax(&ye[q], y);
        atomicAdd(&cnt[q], 1);
    }
}

__global__ void bbox_final_kernel(const int* xs, const int* ys, const int* xe,
                                  const int* ye, const int* cnt,
                                  float* bbox, float* valid)
{
    int i = threadIdx.x;
    if (i < 18) {
        int b = i / 9, c = (i % 9) + 1;
        float* r = bbox + (size_t)i * 6;
        r[0] = (float)b; r[1] = (float)xs[i]; r[2] = (float)ys[i];
        r[3] = (float)xe[i]; r[4] = (float)ye[i]; r[5] = (float)c;
        valid[i] = (cnt[i] >= 100) ? 1.f : 0.f;
    }
}

// ---------------------------------------------------------------------------
// ROI max-pool (f1 @ 1/8 + f2 @ 1/16), NHWC f16 feats -> f32 16x512x7x7
// ---------------------------------------------------------------------------
__device__ __forceinline__ float roi_one(const _Float16* __restrict__ feat, int b, int c,
                                         int Hf, int Wf, float scale,
                                         int bx1, int by1, int bx2, int by2, int i, int j)
{
    float x1 = rintf((float)bx1 * scale), y1 = rintf((float)by1 * scale);
    float x2 = rintf((float)bx2 * scale), y2 = rintf((float)by2 * scale);
    float bw = fmaxf(x2 - x1 + 1.f, 1.f) * (1.f / 7.f);
    float bh = fmaxf(y2 - y1 + 1.f, 1.f) * (1.f / 7.f);
    int ws = (int)fminf(fmaxf(floorf(j * bw) + x1, 0.f), (float)Wf);
    int we = (int)fminf(fmaxf(ceilf((j + 1) * bw) + x1, 0.f), (float)Wf);
    int hs = (int)fminf(fmaxf(floorf(i * bh) + y1, 0.f), (float)Hf);
    int he = (int)fminf(fmaxf(ceilf((i + 1) * bh) + y1, 0.f), (float)Hf);
    float m = NEGV;
    const _Float16* p = feat + (size_t)b * Hf * Wf * 512 + c;
    for (int yy = hs; yy < he; ++yy)
        for (int xx = ws; xx < we; ++xx)
            m = fmaxf(m, (float)p[((size_t)yy * Wf + xx) * 512]);
    return (m <= NEGV * 0.5f) ? 0.f : m;
}

__global__ void roi_pool_kernel(const _Float16* __restrict__ f1,
                                const _Float16* __restrict__ f2,
                                const int* __restrict__ bbx, float* __restrict__ out, int total)
{
    int idx = blockIdx.x * blockDim.x + threadIdx.x;
    if (idx >= total) return;   // total = 16*512*49
    int j = idx % 7;  int t = idx / 7;
    int i = t % 7;    t /= 7;
    int c = t % 512;  int r = t / 512;
    int b   = bbx[r * 5 + 0];
    int bx1 = bbx[r * 5 + 1], by1 = bbx[r * 5 + 2];
    int bx2 = bbx[r * 5 + 3], by2 = bbx[r * 5 + 4];
    float v1 = roi_one(f1, b, c, 60, 80, 0.125f,  bx1, by1, bx2, by2, i, j);
    float v2 = roi_one(f2, b, c, 30, 40, 0.0625f, bx1, by1, bx2, by2, i, j);
    out[(size_t)r * 25088 + (size_t)c * 49 + i * 7 + j] = v1 + v2;
}

// ---------------------------------------------------------------------------
// Host-side orchestration
// ---------------------------------------------------------------------------
extern "C" void kernel_launch(void* const* d_in, const int* in_sizes, int n_in,
                              void* d_out, int out_size, void* d_ws, size_t ws_size,
                              hipStream_t stream)
{
    (void)in_sizes; (void)n_in; (void)out_size; (void)ws_size;
    const int Bn = 2, Himg = 480, Wimg = 640;
    const size_t HW = (size_t)Himg * Wimg;

    // Input indices (setup_inputs dict order, lists expanded):
    // 0 rgb, 1 bbx, 2..14 vgg_w, 15..27 vgg_b, 28..33 seg, 34..39 tr, 40..45 fc
    const float* rgb = (const float*)d_in[0];
    const int*   bbx = (const int*)d_in[1];

    // Output layout (flat, return order)
    float* outf = (float*)d_out;
    size_t off_prob  = 0;
    size_t off_seg   = off_prob  + (size_t)Bn * 11 * HW;
    size_t off_bbox  = off_seg   + (size_t)Bn * HW;
    size_t off_valid = off_bbox  + (size_t)Bn * 9 * 6;
    size_t off_trans = off_valid + (size_t)Bn * 9;
    size_t off_quat  = off_trans + (size_t)Bn * 30 * HW;

    // Workspace carve-up
    char* ws = (char*)d_ws;
    size_t o = 0;
    auto carve = [&](size_t bytes) -> void* {
        void* p = ws + o;
        o += (bytes + 255) & ~(size_t)255;
        return p;
    };
    _Float16* bufA = (_Float16*)carve((size_t)Bn * HW * 64 * 2);   // 78.6 MB
    _Float16* bufB = (_Float16*)carve((size_t)Bn * HW * 64 * 2);   // 78.6 MB
    _Float16* f1   = (_Float16*)carve((size_t)Bn * 60 * 80 * 512 * 2);
    _Float16* f2   = (_Float16*)carve((size_t)Bn * 30 * 40 * 512 * 2);
    float* segf = (float*)carve((size_t)Bn * 64 * 60 * 80 * 4);
    float* trf  = (float*)carve((size_t)Bn * 128 * 60 * 80 * 4);
    float* tmps = (float*)carve((size_t)Bn * 128 * 30 * 40 * 4);
    float* roib = (float*)carve((size_t)16 * 25088 * 4);
    float* fcb1 = (float*)carve((size_t)16 * 4096 * 4);
    float* fcb2 = (float*)carve((size_t)16 * 4096 * 4);
    int*   ibuf = (int*)carve(4096);
    int *xs = ibuf, *ysv = ibuf + 18, *xe = ibuf + 36, *ye = ibuf + 54, *cnt = ibuf + 72;

    auto conv = [&](const _Float16* in, _Float16* out, int widx,
                    int Cin, int Cpad, int Cout, int H_, int W_) {
        dim3 g((W_ + 31) / 32, H_, Bn * ((Cout + 127) / 128));
        conv3x3_wmma_kernel<<<g, 256, 0, stream>>>(
            in, (const float*)d_in[2 + widx], (const float*)d_in[15 + widx], out,
            Cin, Cpad, Cout, H_, W_);
    };
    auto pool = [&](const _Float16* in, _Float16* out, int C, int H_, int W_) {
        int total = Bn * (H_ / 2) * (W_ / 2) * C;
        maxpool2_nhwc_kernel<<<(total + 255) / 256, 256, 0, stream>>>(
            in, out, C, H_, W_, total);
    };

    // ---- input conversion: NCHW f32 -> NHWC f16, 3 channels padded to 32 ----
    {
        int tp = Bn * Himg * Wimg;
        nchw2nhwc_pad_kernel<<<(tp + 255) / 256, 256, 0, stream>>>(
            rgb, bufA, 3, 32, Himg, Wimg, tp);
    }

    // ---- VGG backbone (NHWC f16 everywhere) ----
    conv(bufA, bufB, 0,   3, 32,  64, 480, 640);
    conv(bufB, bufA, 1,  64, 64,  64, 480, 640);
    pool(bufA, bufB, 64, 480, 640);
    conv(bufB, bufA, 2,  64, 64, 128, 240, 320);
    conv(bufA, bufB, 3, 128, 128, 128, 240, 320);
    pool(bufB, bufA, 128, 240, 320);
    conv(bufA, bufB, 4, 128, 128, 256, 120, 160);
    conv(bufB, bufA, 5, 256, 256, 256, 120, 160);
    conv(bufA, bufB, 6, 256, 256, 256, 120, 160);
    pool(bufB, bufA, 256, 120, 160);
    conv(bufA, bufB, 7, 256, 256, 512, 60, 80);
    conv(bufB, bufA, 8, 512, 512, 512, 60, 80);
    conv(bufA, f1,   9, 512, 512, 512, 60, 80);     // f1: B x 60 x 80 x 512
    pool(f1, bufA, 512, 60, 80);
    conv(bufA, bufB, 10, 512, 512, 512, 30, 40);
    conv(bufB, bufA, 11, 512, 512, 512, 30, 40);
    conv(bufA, f2,   12, 512, 512, 512, 30, 40);    // f2: B x 30 x 40 x 512

    // ---- segmentation branch ----
    {
        int t1 = Bn * 64 * 60 * 80;
        conv1x1_relu_kernel<<<(t1 + 255) / 256, 256, 0, stream>>>(
            f1, (const float*)d_in[28], (const float*)d_in[29], segf, 512, 64, 60 * 80, t1);
        int t2 = Bn * 64 * 30 * 40;
        conv1x1_relu_kernel<<<(t2 + 255) / 256, 256, 0, stream>>>(
            f2, (const float*)d_in[30], (const float*)d_in[31], tmps, 512, 64, 30 * 40, t2);
        up2add_kernel<<<(t1 + 255) / 256, 256, 0, stream>>>(tmps, segf, 64, 30, 40, t1);
        int tp = Bn * Himg * Wimg;
        seg_head_kernel<<<(tp + 255) / 256, 256, 0, stream>>>(
            segf, (const float*)d_in[32], (const float*)d_in[33],
            outf + off_prob, outf + off_seg, tp);
    }

    // ---- bounding boxes from seg ----
    {
        bbox_init_kernel<<<1, 32, 0, stream>>>(xs, ysv, xe, ye, cnt);
        int tp = Bn * Himg * Wimg;
        bbox_scan_kernel<<<(tp + 255) / 256, 256, 0, stream>>>(
            outf + off_seg, xs, ysv, xe, ye, cnt, tp);
        bbox_final_kernel<<<1, 32, 0, stream>>>(xs, ysv, xe, ye, cnt,
                                                outf + off_bbox, outf + off_valid);
    }

    // ---- translation branch ----
    {
        int t1 = Bn * 128 * 60 * 80;
        conv1x1_relu_kernel<<<(t1 + 255) / 256, 256, 0, stream>>>(
            f1, (const float*)d_in[34], (const float*)d_in[35], trf, 512, 128, 60 * 80, t1);
        int t2 = Bn * 128 * 30 * 40;
        conv1x1_relu_kernel<<<(t2 + 255) / 256, 256, 0, stream>>>(
            f2, (const float*)d_in[36], (const float*)d_in[37], tmps, 512, 128, 30 * 40, t2);
        up2add_kernel<<<(t1 + 255) / 256, 256, 0, stream>>>(tmps, trf, 128, 30, 40, t1);
        int tp = Bn * Himg * Wimg;
        tr_head_kernel<<<(tp + 255) / 256, 256, 0, stream>>>(
            trf, (const float*)d_in[38], (const float*)d_in[39], outf + off_trans, tp);
    }

    // ---- ROI pooling + FC head (WMMA GEMMs) ----
    {
        int tr_ = 16 * 512 * 49;
        roi_pool_kernel<<<(tr_ + 255) / 256, 256, 0, stream>>>(f1, f2, bbx, roib, tr_);
        fc_wmma_kernel<<<(4096 + 15) / 16, 32, 0, stream>>>(
            roib, (const float*)d_in[40], (const float*)d_in[41], fcb1,
            16, 4096, 25088, 1);
        fc_wmma_kernel<<<(4096 + 15) / 16, 32, 0, stream>>>(
            fcb1, (const float*)d_in[42], (const float*)d_in[43], fcb2,
            16, 4096, 4096, 1);
        fc_wmma_kernel<<<(40 + 15) / 16, 32, 0, stream>>>(
            fcb2, (const float*)d_in[44], (const float*)d_in[45], outf + off_quat,
            16, 40, 4096, 0);
    }
}
